// PyTorchAttentionMechanism_50964081935147
// MI455X (gfx1250) — matrix operand order repaired and checked by reading, not compiled, not run on hardware
//
#include <hip/hip_runtime.h>
#include <hip/hip_bf16.h>

// ---------- vector types for WMMA fragments / TDM descriptors ----------
typedef _Float16 h8  __attribute__((ext_vector_type(8)));
typedef _Float16 h16 __attribute__((ext_vector_type(16)));
typedef float    f8  __attribute__((ext_vector_type(8)));
typedef unsigned int u32x4 __attribute__((ext_vector_type(4)));
typedef int          i32x4 __attribute__((ext_vector_type(4)));
typedef int          i32x8 __attribute__((ext_vector_type(8)));

union H16U { h16 v; h8 h[2]; };

static __device__ __forceinline__ h16 ld_frag(const _Float16* p0, const _Float16* p1) {
    H16U u;
    u.h[0] = *(const h8*)p0;
    u.h[1] = *(const h8*)p1;
    return u.v;
}

static __device__ __forceinline__ f8 wmma_f16(h16 a, h16 b, f8 c) {
    // D = A(16x32 f16) x B(32x16 f16) + C(16x16 f32)
    return __builtin_amdgcn_wmma_f32_16x16x32_f16(
        /*neg_a=*/false, a, /*neg_b=*/false, b,
        /*c_mod=*/(short)0, c, /*reuse_a=*/false, /*reuse_b=*/false);
}

// DPP16 lane rotation within each 16-lane row (VALU, no DS-pipe traffic).
// CTRL = 0x120 | n  -> row_ror:n
template <int CTRL>
static __device__ __forceinline__ float dpp_rot(float x) {
    return __builtin_bit_cast(float, __builtin_amdgcn_update_dpp(
        0, __builtin_bit_cast(int, x), CTRL, 0xF, 0xF, true));
}
// Full 16-lane-row reductions via rotate butterflies (masks 1,2,4,8 over Z/16)
static __device__ __forceinline__ float row_max16(float x) {
    x = fmaxf(x, dpp_rot<0x121>(x));
    x = fmaxf(x, dpp_rot<0x122>(x));
    x = fmaxf(x, dpp_rot<0x124>(x));
    x = fmaxf(x, dpp_rot<0x128>(x));
    return x;
}
static __device__ __forceinline__ float row_sum16(float x) {
    x += dpp_rot<0x121>(x);
    x += dpp_rot<0x122>(x);
    x += dpp_rot<0x124>(x);
    x += dpp_rot<0x128>(x);
    return x;
}

// ---------------------------------------------------------------------------
// Tiled WMMA GEMM:  C[M=4096, N=1024] = A[M,1024] * W[1024,1024] (+bias)*scale
// Block tile 128x128, K-step 32. 8 waves, each wave owns a 32x64 sub-tile.
// MODE 0: store f16 head-split [B,H,S,64]   (QKV projections)
// MODE 1: store f32 row-major [M,N]         (output projection -> d_out)
// ---------------------------------------------------------------------------
template <typename AT, int MODE>
__global__ __launch_bounds__(256)
void gemm_wmma_kernel(const AT* __restrict__ A, const float* __restrict__ W,
                      const float* __restrict__ bias, void* __restrict__ out,
                      float scale) {
    constexpr int Kd = 1024, Nd = 1024;
    __shared__ _Float16 Alds[128 * 32];   // [m][k] row-major
    __shared__ _Float16 Bt[128 * 32];     // [n][k] (W transposed tile)

    const int tid  = threadIdx.x;
    const int lane = tid & 31;
    const int w    = tid >> 5;
    const int wm   = (w & 3) * 32;        // wave M origin within block
    const int wn   = (w >> 2) * 64;       // wave N origin within block
    const int m0   = blockIdx.y * 128;
    const int n0   = blockIdx.x * 128;
    const int col  = lane & 15;
    const int hf   = lane >> 4;
    const int kb   = hf * 8;              // A-fragment K base
    const int ks   = hf * 16;             // unused; B fragments also use kb

    f8 acc[2][4];
#pragma unroll
    for (int i = 0; i < 2; i++)
#pragma unroll
        for (int j = 0; j < 4; j++) acc[i][j] = {};

    for (int k0 = 0; k0 < Kd; k0 += 32) {
        __syncthreads();
        // Stage A tile 128x32 (convert to f16). Coalesced over k.
#pragma unroll
        for (int i = 0; i < 16; i++) {
            int idx = i * 256 + tid;             // == m*32 + k
            int mm = idx >> 5, kk = idx & 31;
            Alds[idx] = (_Float16)A[(size_t)(m0 + mm) * Kd + k0 + kk];
        }
        // Stage W tile 32x128 transposed -> Bt[n][k]. Coalesced over n.
#pragma unroll
        for (int i = 0; i < 16; i++) {
            int idx = i * 256 + tid;
            int kk = idx >> 7, nn = idx & 127;
            Bt[nn * 32 + kk] = (_Float16)W[(size_t)(k0 + kk) * Nd + n0 + nn];
        }
        __syncthreads();

        // A fragments: lane holds row (lane&15), K chunk per ISA 16-bit A layout
        h16 a0 = ld_frag(&Alds[(wm +       col) * 32 + kb],
                         &Alds[(wm +       col) * 32 + kb + 16]);
        h16 a1 = ld_frag(&Alds[(wm + 16 +  col) * 32 + kb],
                         &Alds[(wm + 16 +  col) * 32 + kb + 16]);
#pragma unroll
        for (int j = 0; j < 4; j++) {
            const int bc = wn + j * 16 + col;    // B column (N)
            h16 b = ld_frag(&Bt[bc * 32 + kb], &Bt[bc * 32 + kb + 16]);
            acc[0][j] = wmma_f16(a0, b, acc[0][j]);
            acc[1][j] = wmma_f16(a1, b, acc[1][j]);
        }
    }

    // Store. C layout: element (M = v + 8*hf, N = lane&15) in VGPR v.
#pragma unroll
    for (int i = 0; i < 2; i++) {
#pragma unroll
        for (int j = 0; j < 4; j++) {
            const int nG = n0 + wn + j * 16 + col;
            const float bs = bias[nG];
#pragma unroll
            for (int v = 0; v < 8; v++) {
                const int mG = m0 + wm + i * 16 + v + 8 * hf;
                const float val = (acc[i][j][v] + bs) * scale;
                if (MODE == 0) {
                    // head-split f16 [B,H,S,64]; m = b*2048+s, n = h*64+hd
                    const int bb = mG >> 11, ss = mG & 2047;
                    const int hh = nG >> 6,  hd = nG & 63;
                    ((_Float16*)out)[(((size_t)(bb * 16 + hh) * 2048) + ss) * 64 + hd] =
                        (_Float16)val;
                } else {
                    ((float*)out)[(size_t)mG * Nd + nG] = val;
                }
            }
        }
    }
}

// Issue a TDM 64x64 f16 tile load (contiguous rows) into LDS. Wave-uniform args.
static __device__ __forceinline__ void tdm_load_64x64_f16(unsigned lds_off,
                                                          const _Float16* gptr) {
    const unsigned long long ga = (unsigned long long)(uintptr_t)gptr;
    u32x4 g0;
    g0.x = 0x1u;                                   // count=1, user mode
    g0.y = lds_off;                                // lds_addr
    g0.z = (unsigned)(ga & 0xffffffffu);           // global_addr[31:0]
    g0.w = (unsigned)((ga >> 32) & 0x01ffffffu)    // global_addr[56:32]
         | (2u << 30);                             // type=2 (image/D#)
    i32x8 g1 = { 0x00010000,        // workgroup_mask=0 | data_size=1(2B)
                 (64 << 16),        // tensor_dim0=64 (bits 79:48, low part)
                 (64 << 16),        // tensor_dim0 hi=0 | tensor_dim1=64 lo
                 (64 << 16),        // tensor_dim1 hi=0 | tile_dim0=64
                 64,                // tile_dim1=64, tile_dim2=0
                 64,                // tensor_dim0_stride=64 (low 32)
                 0, 0 };            // stride hi / tensor_dim1_stride=0
    i32x4 gz  = { 0, 0, 0, 0 };     // 2-D tensor: groups 2/3 unused
    i32x8 gz8 = { 0, 0, 0, 0, 0, 0, 0, 0 };
    __builtin_amdgcn_tensor_load_to_lds(g0, g1, gz, gz, gz8, 0);
}

// ---------------------------------------------------------------------------
// Flash attention: grid (S/128, H, B), 256 threads (8 waves).
// Each wave owns 16 query rows; loop over 64-key blocks with online softmax
// kept entirely in registers (DPP row reductions, no DS-pipe shuffles).
// K blocks are double-buffered through the Tensor Data Mover: wave 0 issues
// the DMA for block i+1 at the top of iteration i and waits TENSORcnt only
// at the bottom, so the transfer overlaps the 16 WMMAs + softmax of block i.
// ---------------------------------------------------------------------------
__global__ __launch_bounds__(256)
void attn_wmma_kernel(const _Float16* __restrict__ Q, const _Float16* __restrict__ Kk,
                      const _Float16* __restrict__ V, _Float16* __restrict__ ctx) {
    constexpr int S = 2048, HD = 64, H = 16;
    __shared__ _Float16 Klds[2][64 * 64];     // ping-pong [key][hd] (TDM dest)
    __shared__ _Float16 Vt[64 * 64];          // [hd][key] transposed
    __shared__ _Float16 Plds[8][16 * 64];     // per-wave P tile (f16)

    const int tid  = threadIdx.x;
    const int lane = tid & 31;
    const int w    = tid >> 5;
    const int col  = lane & 15;
    const int hf   = lane >> 4;
    const int kb   = hf * 8;
    const int ks   = hf * 16;
    const int q0   = blockIdx.x * 128;
    const int h    = blockIdx.y;
    const int b    = blockIdx.z;
    const size_t base = (size_t)(b * H + h) * S * HD;
    const _Float16* kbase_p = Kk + base;

    // LDS byte offsets: generic LDS pointers carry the offset in addr[31:0]
    const unsigned klds_off0 = (unsigned)(unsigned long long)(uintptr_t)&Klds[0][0];
    const unsigned klds_off1 = (unsigned)(unsigned long long)(uintptr_t)&Klds[1][0];

    // Q fragments for this wave's 16 rows (Q already scaled by 1/sqrt(64))
    const int qrow = q0 + w * 16 + col;
    const _Float16* qp = Q + base + (size_t)qrow * HD;
    h16 qa0 = ld_frag(qp + kb,      qp + kb + 16);       // hd 0..31
    h16 qa1 = ld_frag(qp + 32 + kb, qp + 32 + kb + 16);  // hd 32..63

    f8 acc[4];
#pragma unroll
    for (int n = 0; n < 4; n++) acc[n] = {};
    float m_i[8], l_i[8];
#pragma unroll
    for (int v = 0; v < 8; v++) { m_i[v] = -1e30f; l_i[v] = 0.0f; }

    // Prologue: DMA key block 0 into buffer 0 (wave 0 issues + waits)
    if (w == 0) {
        tdm_load_64x64_f16(klds_off0, kbase_p);
        __builtin_amdgcn_s_wait_tensorcnt((short)0);
    }

    for (int kb0 = 0; kb0 < S; kb0 += 64) {
        const int cur = (kb0 >> 6) & 1;
        __syncthreads();   // block i-1 compute done; kbuf[cur] DMA complete

        // Prefetch next K block into the other buffer (overlaps this compute)
        if (w == 0 && (kb0 + 64) < S) {
            tdm_load_64x64_f16(cur ? klds_off0 : klds_off1,
                               kbase_p + (size_t)(kb0 + 64) * HD);
        }

        // Stage V block transposed -> Vt[hd][key]
        const _Float16* vp = V + base + (size_t)kb0 * HD;
#pragma unroll
        for (int i = 0; i < 16; i++) {
            int idx = i * 256 + tid;
            int key = idx >> 6, hd = idx & 63;
            Vt[hd * 64 + key] = vp[idx];
        }
        __syncthreads();   // Vt visible to all waves

        const _Float16* kcur = &Klds[cur][0];

        // scores S = Q * K^T : B fragment column = key, halves = hd (contig rows)
        f8 sc[4];
#pragma unroll
        for (int n = 0; n < 4; n++) {
            const _Float16* krow = kcur + (n * 16 + col) * 64;
            h16 b0 = ld_frag(krow + kb,      krow + kb + 16);       // hd 0..31
            h16 b1 = ld_frag(krow + 32 + kb, krow + 32 + kb + 16);  // hd 32..63
            f8 z = {};
            z = wmma_f16(qa0, b0, z);
            z = wmma_f16(qa1, b1, z);
            sc[n] = z;
        }

        // Online softmax: lane owns rows v + 8*hf; DPP row_ror butterflies
#pragma unroll
        for (int v = 0; v < 8; v++) {
            float mx = fmaxf(fmaxf(sc[0][v], sc[1][v]), fmaxf(sc[2][v], sc[3][v]));
            mx = row_max16(mx);
            const float mnew = fmaxf(m_i[v], mx);
            const float csc  = __expf(m_i[v] - mnew);
            float sum = 0.0f;
            const int prow = (v + 8 * hf) * 64;
#pragma unroll
            for (int n = 0; n < 4; n++) {
                float p = __expf(sc[n][v] - mnew);
                sum += p;
                Plds[w][prow + n * 16 + col] = (_Float16)p;
            }
            sum = row_sum16(sum);
            l_i[v] = l_i[v] * csc + sum;
            m_i[v] = mnew;
#pragma unroll
            for (int n = 0; n < 4; n++) acc[n][v] *= csc;
        }

        // ctx += P * V : A = P (16x64 over keys), B = V (keys x hd), Vt contiguous
        const _Float16* prow = &Plds[w][col * 64];
        h16 pa0 = ld_frag(prow + kb,      prow + kb + 16);       // keys 0..31
        h16 pa1 = ld_frag(prow + 32 + kb, prow + 32 + kb + 16);  // keys 32..63
#pragma unroll
        for (int n = 0; n < 4; n++) {
            const _Float16* vrow = &Vt[(n * 16 + col) * 64];
            h16 b0 = ld_frag(vrow + kb,      vrow + kb + 16);       // keys 0..31
            h16 b1 = ld_frag(vrow + 32 + kb, vrow + 32 + kb + 16);  // keys 32..63
            acc[n] = wmma_f16(pa0, b0, acc[n]);
            acc[n] = wmma_f16(pa1, b1, acc[n]);
        }

        // Ensure the prefetched K block has landed before the next iteration
        if (w == 0) __builtin_amdgcn_s_wait_tensorcnt((short)0);
    }

    // Normalize and store ctx in [B,S,D] layout (f16, feeds output projection)
#pragma unroll
    for (int v = 0; v < 8; v++) {
        const float inv = 1.0f / l_i[v];
        const int qg = q0 + w * 16 + v + 8 * hf;
        const size_t rowbase = ((size_t)b * S + qg) * 1024 + h * 64;
#pragma unroll
        for (int n = 0; n < 4; n++)
            ctx[rowbase + n * 16 + col] = (_Float16)(acc[n][v] * inv);
    }
}

// ---------------------------------------------------------------------------
extern "C" void kernel_launch(void* const* d_in, const int* in_sizes, int n_in,
                              void* d_out, int out_size, void* d_ws, size_t ws_size,
                              hipStream_t stream) {
    const float* query  = (const float*)d_in[0];
    const float* key_in = (const float*)d_in[1];
    const float* value  = (const float*)d_in[2];
    const float* Wq = (const float*)d_in[3];
    const float* bq = (const float*)d_in[4];
    const float* Wk = (const float*)d_in[5];
    const float* bk = (const float*)d_in[6];
    const float* Wv = (const float*)d_in[7];
    const float* bv = (const float*)d_in[8];
    const float* Wo = (const float*)d_in[9];
    const float* bo = (const float*)d_in[10];

    // workspace: Q,K,V head-split f16 (8MB each) + ctx f16 (8MB)
    const size_t QKV_ELTS = (size_t)2 * 16 * 2048 * 64;   // B*H*S*HD
    char* ws = (char*)d_ws;
    _Float16* qws   = (_Float16*)(ws);
    _Float16* kws   = (_Float16*)(ws + QKV_ELTS * 2);
    _Float16* vws   = (_Float16*)(ws + QKV_ELTS * 4);
    _Float16* ctxws = (_Float16*)(ws + QKV_ELTS * 6);

    dim3 ggrid(1024 / 128, 4096 / 128);   // (N/128, M/128)
    const float inv_sqrt_hd = 0.125f;     // 1/sqrt(64) folded into Q

    gemm_wmma_kernel<float, 0><<<ggrid, 256, 0, stream>>>(query,  Wq, bq, qws, inv_sqrt_hd);
    gemm_wmma_kernel<float, 0><<<ggrid, 256, 0, stream>>>(key_in, Wk, bk, kws, 1.0f);
    gemm_wmma_kernel<float, 0><<<ggrid, 256, 0, stream>>>(value,  Wv, bv, vws, 1.0f);

    attn_wmma_kernel<<<dim3(2048 / 128, 16, 2), 256, 0, stream>>>(qws, kws, vws, ctxws);

    gemm_wmma_kernel<_Float16, 1><<<ggrid, 256, 0, stream>>>(ctxws, Wo, bo, d_out, 1.0f);
}
